// VitsAttention_30073361007314
// MI455X (gfx1250) — matrix operand order repaired and checked by reading, not compiled, not run on hardware
//
#include <hip/hip_runtime.h>
#include <hip/hip_bf16.h>
#include <math.h>

typedef __attribute__((ext_vector_type(16))) _Float16 v16h;
typedef __attribute__((ext_vector_type(8)))  _Float16 v8h;
typedef __attribute__((ext_vector_type(8)))  float    v8f;

// ---------------------------------------------------------------------------
// Shapes: B=4, L=1024, E=1024, H=16, D=64, WINDOW=4, SCALING = 1/8
// ---------------------------------------------------------------------------
#define LSEQ 1024
#define EDIM 1024
#define HDS  16
#define DDIM 64
#define BH   64           // B*H
#define MTOT 4096         // B*L

// Build a 16x16x32 f16 A-operand fragment from a row-major 32-half window.
// A layout: lane m = l&15, k = (l>>4)*16 + e  -> contiguous in k.
__device__ inline v16h load_a_frag(const _Float16* row, int hi) {
  union { v16h v; v8h h[2]; } u;
  u.h[0] = *(const v8h*)(row + hi * 16);
  u.h[1] = *(const v8h*)(row + hi * 16 + 8);
  return u.v;
}

// Build a 16x16x32 f16 B-operand fragment from 16 contiguous halves.
// B layout: lane n = l&15, k = (l>>4)*16 + e  -> contiguous in k.
__device__ inline v16h load_b_frag(const _Float16* p) {
  union { v16h v; v8h h[2]; } u;
  u.h[0] = *(const v8h*)(p);
  u.h[1] = *(const v8h*)(p + 8);
  return u.v;
}

__device__ inline v8f wmma_f16(v16h a, v16h b, v8f c) {
  return __builtin_amdgcn_wmma_f32_16x16x32_f16(false, a, false, b, (short)0, c,
                                                false, false);
}

// ---------------------------------------------------------------------------
// One-time pre-pass 1: f32 -> f16 bulk convert (hidden_states).
// ---------------------------------------------------------------------------
__global__ __launch_bounds__(256) void cvt_f32_f16(const float* __restrict__ X,
                                                   _Float16* __restrict__ Y,
                                                   int n8) {
  const int i = blockIdx.x * 256 + threadIdx.x;
  if (i >= n8) return;
  const float4 a = ((const float4*)X)[2 * i + 0];
  const float4 b = ((const float4*)X)[2 * i + 1];
  v8h o;
  o[0] = (_Float16)a.x; o[1] = (_Float16)a.y;
  o[2] = (_Float16)a.z; o[3] = (_Float16)a.w;
  o[4] = (_Float16)b.x; o[5] = (_Float16)b.y;
  o[6] = (_Float16)b.z; o[7] = (_Float16)b.w;
  ((v8h*)Y)[i] = o;
}

// ---------------------------------------------------------------------------
// One-time pre-pass 2: W (1024x1024 f32, k-major) -> Wt (1024x1024 f16,
// n-major) so B fragments are contiguous-in-k global loads. 64x64 LDS tiles.
// ---------------------------------------------------------------------------
__global__ __launch_bounds__(256) void transpose_w(const float* __restrict__ W,
                                                   _Float16* __restrict__ Wt) {
  __shared__ _Float16 T[64][72];        // [n_local][k_local], padded
  const int bn = (blockIdx.x & 15) * 64;   // n tile
  const int bk = (blockIdx.x >> 4) * 64;   // k tile
  const int tid = threadIdx.x;
#pragma unroll
  for (int i = 0; i < 4; ++i) {         // 64 rows x 16 float4 = 1024 slots
    int slot = tid + 256 * i;
    int r = slot >> 4, c4 = (slot & 15) * 4;
    float4 d = *(const float4*)(W + (size_t)(bk + r) * EDIM + bn + c4);
    T[c4 + 0][r] = (_Float16)d.x;
    T[c4 + 1][r] = (_Float16)d.y;
    T[c4 + 2][r] = (_Float16)d.z;
    T[c4 + 3][r] = (_Float16)d.w;
  }
  __syncthreads();
#pragma unroll
  for (int i = 0; i < 2; ++i) {         // 64 rows x 8 v8h = 512 slots
    int slot = tid + 256 * i;
    int r = slot >> 3, c8 = (slot & 7) * 8;
    v8h o;
#pragma unroll
    for (int j = 0; j < 8; ++j) o[j] = T[r][c8 + j];
    *(v8h*)(Wt + (size_t)(bn + r) * EDIM + bk + c8) = o;
  }
}

// ---------------------------------------------------------------------------
// LDS-free WMMA GEMM: C(4096 x 1024) = A_f16 @ Wt_f16^T(n-major) + bias.
// Operands fetched straight from global in contiguous-k (k = 16*hi + e) WMMA lane layout; the
// 192MB L2 absorbs cross-block re-reads. 8 waves x (32x64) per 128x128 block.
// out_mode: 0 = f32 row-major, 1 = f16 per-head (BH,L,D), 2 = f16 (BH,D,L).
// ---------------------------------------------------------------------------
__global__ __launch_bounds__(256) void gemm_wmma(const _Float16* __restrict__ A,
                                                 const _Float16* __restrict__ Wt,
                                                 const float* __restrict__ bias,
                                                 void* __restrict__ Out,
                                                 int out_mode, float scale) {
  const int lane = threadIdx.x & 31;
  const int wid  = threadIdx.x >> 5;
  const int hi   = lane >> 4;
  const int n16  = lane & 15;

  const int bm = (blockIdx.x >> 3) * 128;
  const int bn = (blockIdx.x & 7) * 128;
  const int wm = bm + (wid >> 1) * 32;   // 4 waves along M
  const int wn = bn + (wid & 1) * 64;    // 2 waves along N

  const _Float16* arow0 = A + (size_t)(wm + n16) * EDIM;
  const _Float16* arow1 = A + (size_t)(wm + 16 + n16) * EDIM;
  const _Float16* brow0 = Wt + (size_t)(wn + n16) * EDIM + hi * 16;
  const _Float16* brow1 = brow0 + (size_t)16 * EDIM;
  const _Float16* brow2 = brow0 + (size_t)32 * EDIM;
  const _Float16* brow3 = brow0 + (size_t)48 * EDIM;

  v8f acc[2][4] = {};
  for (int kb = 0; kb < EDIM; kb += 32) {
    const v16h a0 = load_a_frag(arow0 + kb, hi);
    const v16h a1 = load_a_frag(arow1 + kb, hi);
    const v16h b0 = load_b_frag(brow0 + kb);
    const v16h b1 = load_b_frag(brow1 + kb);
    const v16h b2 = load_b_frag(brow2 + kb);
    const v16h b3 = load_b_frag(brow3 + kb);
    acc[0][0] = wmma_f16(a0, b0, acc[0][0]);
    acc[0][1] = wmma_f16(a0, b1, acc[0][1]);
    acc[0][2] = wmma_f16(a0, b2, acc[0][2]);
    acc[0][3] = wmma_f16(a0, b3, acc[0][3]);
    acc[1][0] = wmma_f16(a1, b0, acc[1][0]);
    acc[1][1] = wmma_f16(a1, b1, acc[1][1]);
    acc[1][2] = wmma_f16(a1, b2, acc[1][2]);
    acc[1][3] = wmma_f16(a1, b3, acc[1][3]);
  }

  // ---- epilogue
#pragma unroll
  for (int nt = 0; nt < 4; ++nt) {
    const int n_g = wn + nt * 16 + n16;
    const float bv = bias[n_g];
#pragma unroll
    for (int mt = 0; mt < 2; ++mt) {
#pragma unroll
      for (int v = 0; v < 8; ++v) {
        const int m_g = wm + mt * 16 + v + hi * 8;
        const float val = (acc[mt][nt][v] + bv) * scale;
        if (out_mode == 0) {
          ((float*)Out)[(size_t)m_g * EDIM + n_g] = val;
        } else {
          const int b = m_g >> 10, t = m_g & 1023;
          const int h = n_g >> 6, d = n_g & 63;
          _Float16* o16 = (_Float16*)Out;
          if (out_mode == 1)
            o16[(((size_t)(b * HDS + h)) * LSEQ + t) * DDIM + d] = (_Float16)val;
          else
            o16[(((size_t)(b * HDS + h)) * DDIM + d) * LSEQ + t] = (_Float16)val;
        }
      }
    }
  }
}

// ---------------------------------------------------------------------------
// Banded rel-k logits: relk[bh, t, r] = q[bh,t,:] . emb_rel_k[r,:], r=0..8
// (the dense (BH,L,2L-1) rel tensor is zero outside this 9-wide band)
// ---------------------------------------------------------------------------
__global__ __launch_bounds__(256) void relk_kernel(const _Float16* __restrict__ Qf,
                                                   const float* __restrict__ erk,
                                                   float* __restrict__ relk) {
  const int idx = blockIdx.x * 256 + threadIdx.x;   // BH*L = 65536
  if (idx >= BH * LSEQ) return;
  const _Float16* q = Qf + (size_t)idx * DDIM;
  float qv[DDIM];
#pragma unroll
  for (int d = 0; d < DDIM; ++d) qv[d] = (float)q[d];
#pragma unroll
  for (int r = 0; r < 9; ++r) {
    const float* e = erk + r * DDIM;
    float s = 0.f;
#pragma unroll
    for (int d = 0; d < DDIM; ++d) s += qv[d] * e[d];
    relk[(size_t)idx * 9 + r] = s;
  }
}

// ---------------------------------------------------------------------------
// Flash attention with banded relative bias.
// One wave per (head, 16-row tile). 8 waves / block -> 512 blocks.
// ---------------------------------------------------------------------------
__global__ __launch_bounds__(256) void attn_kernel(const _Float16* __restrict__ Qf,
                                                   const _Float16* __restrict__ Kf,
                                                   const _Float16* __restrict__ Vt,
                                                   const float* __restrict__ relk,
                                                   const float* __restrict__ erv,
                                                   _Float16* __restrict__ Xatt) {
  __shared__ _Float16 pbuf[8][16][32];   // per-wave P staging (C-layout -> A-layout)
  __shared__ float    band[8][16][9];    // per-wave banded logits S[i, i+r-4]
  __shared__ float    relv[9][DDIM];

  const int tid  = threadIdx.x;
  const int lane = tid & 31;
  const int wid  = tid >> 5;
  const int hi   = lane >> 4;
  const int n16  = lane & 15;

  for (int i = tid; i < 9 * DDIM; i += 256) relv[i >> 6][i & 63] = erv[i];
  for (int i = lane; i < 144; i += 32) band[wid][i / 9][i % 9] = -INFINITY;
  __syncthreads();

  const int task   = blockIdx.x * 8 + wid;     // 0..4095
  const int bh     = task >> 6;                // 0..63
  const int m_base = (task & 63) * 16;

  // Q fragment: rows m_base..m_base+15, k split 0..31 / 32..63
  const _Float16* qrow = Qf + ((size_t)bh * LSEQ + m_base + n16) * DDIM;
  const v16h a0 = load_a_frag(qrow, hi);
  const v16h a1 = load_a_frag(qrow + 32, hi);

  float rmax[8], rsum[8];
#pragma unroll
  for (int v = 0; v < 8; ++v) { rmax[v] = -INFINITY; rsum[v] = 0.f; }
  v8f o[4] = {};

  for (int kb = 0; kb < LSEQ; kb += 32) {
    // ---- S = Q . K^T for two 16-wide key tiles
    const _Float16* k0 = Kf + ((size_t)bh * LSEQ + kb + n16) * DDIM + hi * 16;
    const _Float16* k1 = k0 + 16 * DDIM;
    v8f c0 = {}, c1 = {};
    c0 = wmma_f16(a0, load_b_frag(k0), c0);
    c0 = wmma_f16(a1, load_b_frag(k0 + 32), c0);
    c1 = wmma_f16(a0, load_b_frag(k1), c1);
    c1 = wmma_f16(a1, load_b_frag(k1 + 32), c1);

    // ---- banded relative-position bias (only when chunk touches the band)
    if (kb + 31 >= m_base - 4 && kb <= m_base + 19) {
#pragma unroll
      for (int v = 0; v < 8; ++v) {
        const int m_loc = v + hi * 8;
        const int m_g   = m_base + m_loc;
        const int r0 = (kb + n16) - m_g + 4;
        if (r0 >= 0 && r0 <= 8) {
          c0[v] += relk[((size_t)bh * LSEQ + m_g) * 9 + r0];
          band[wid][m_loc][r0] = c0[v];
        }
        const int r1 = (kb + 16 + n16) - m_g + 4;
        if (r1 >= 0 && r1 <= 8) {
          c1[v] += relk[((size_t)bh * LSEQ + m_g) * 9 + r1];
          band[wid][m_loc][r1] = c1[v];
        }
      }
    }

    // ---- online softmax (rows live in 16-lane half-wave groups)
#pragma unroll
    for (int v = 0; v < 8; ++v) {
      float cm = fmaxf(c0[v], c1[v]);
      cm = fmaxf(cm, __shfl_xor(cm, 8, 32));
      cm = fmaxf(cm, __shfl_xor(cm, 4, 32));
      cm = fmaxf(cm, __shfl_xor(cm, 2, 32));
      cm = fmaxf(cm, __shfl_xor(cm, 1, 32));
      const float nm   = fmaxf(rmax[v], cm);
      const float corr = __expf(rmax[v] - nm);
      rmax[v] = nm;
      const float p0 = __expf(c0[v] - nm);
      const float p1 = __expf(c1[v] - nm);
      float ps = p0 + p1;
      ps += __shfl_xor(ps, 8, 32);
      ps += __shfl_xor(ps, 4, 32);
      ps += __shfl_xor(ps, 2, 32);
      ps += __shfl_xor(ps, 1, 32);
      rsum[v] = rsum[v] * corr + ps;
      o[0][v] *= corr; o[1][v] *= corr; o[2][v] *= corr; o[3][v] *= corr;
      const int m_loc = v + hi * 8;
      pbuf[wid][m_loc][n16]      = (_Float16)p0;
      pbuf[wid][m_loc][16 + n16] = (_Float16)p1;
    }
    __syncthreads();

    // ---- O += P . V  (V stored transposed: contiguous B-operand loads)
    const v16h ap = load_a_frag(&pbuf[wid][n16][0], hi);
#pragma unroll
    for (int dt = 0; dt < 4; ++dt) {
      const _Float16* vp =
          Vt + ((size_t)bh * DDIM + dt * 16 + n16) * LSEQ + kb + hi * 16;
      o[dt] = wmma_f16(ap, load_b_frag(vp), o[dt]);
    }
    __syncthreads();
  }

  // ---- epilogue: normalize, add banded rel-v term, store (B, L, E) f16
  const int b = bh >> 4, h = bh & 15;
#pragma unroll
  for (int v = 0; v < 8; ++v) {
    const float inv  = 1.0f / rsum[v];
    const int m_loc  = v + hi * 8;
    float pb[9];
#pragma unroll
    for (int r = 0; r < 9; ++r)
      pb[r] = __expf(band[wid][m_loc][r] - rmax[v]) * inv;
#pragma unroll
    for (int dt = 0; dt < 4; ++dt) {
      const int d = dt * 16 + n16;
      float val = o[dt][v] * inv;
#pragma unroll
      for (int r = 0; r < 9; ++r) val += pb[r] * relv[r][d];
      const int t_g = m_base + m_loc;
      Xatt[((size_t)(b * LSEQ + t_g)) * EDIM + h * DDIM + d] = (_Float16)val;
    }
  }
}

// ---------------------------------------------------------------------------
extern "C" void kernel_launch(void* const* d_in, const int* in_sizes, int n_in,
                              void* d_out, int out_size, void* d_ws, size_t ws_size,
                              hipStream_t stream) {
  const float* hs  = (const float*)d_in[0];
  const float* Wq  = (const float*)d_in[1];
  const float* bq  = (const float*)d_in[2];
  const float* Wk  = (const float*)d_in[3];
  const float* bk  = (const float*)d_in[4];
  const float* Wv  = (const float*)d_in[5];
  const float* bv  = (const float*)d_in[6];
  const float* Wo  = (const float*)d_in[7];
  const float* bo  = (const float*)d_in[8];
  const float* erk = (const float*)d_in[9];
  const float* erv = (const float*)d_in[10];
  float* out = (float*)d_out;

  char* ws = (char*)d_ws;
  const size_t MB = (size_t)1 << 20;
  _Float16* Qf   = (_Float16*)(ws);
  _Float16* Kf   = (_Float16*)(ws + 8 * MB);
  _Float16* Vt   = (_Float16*)(ws + 16 * MB);
  _Float16* Xatt = (_Float16*)(ws + 24 * MB);
  _Float16* Xh   = (_Float16*)(ws + 32 * MB);
  _Float16* Wtq  = (_Float16*)(ws + 40 * MB);
  _Float16* Wtk  = (_Float16*)(ws + 42 * MB);
  _Float16* Wtv  = (_Float16*)(ws + 44 * MB);
  _Float16* Wto  = (_Float16*)(ws + 46 * MB);
  float*    relk = (float*)   (ws + 48 * MB);

  // One-time pre-passes: hidden f32->f16; weights f32 k-major -> f16 n-major.
  cvt_f32_f16<<<(MTOT * EDIM / 8) / 256, 256, 0, stream>>>(hs, Xh, MTOT * EDIM / 8);
  transpose_w<<<256, 256, 0, stream>>>(Wq, Wtq);
  transpose_w<<<256, 256, 0, stream>>>(Wk, Wtk);
  transpose_w<<<256, 256, 0, stream>>>(Wv, Wtv);
  transpose_w<<<256, 256, 0, stream>>>(Wo, Wto);

  // Projections: Q (scaled, per-head), K (per-head), V (per-head transposed)
  gemm_wmma<<<256, 256, 0, stream>>>(Xh, Wtq, bq, Qf, 1, 0.125f);
  gemm_wmma<<<256, 256, 0, stream>>>(Xh, Wtk, bk, Kf, 1, 1.0f);
  gemm_wmma<<<256, 256, 0, stream>>>(Xh, Wtv, bv, Vt, 2, 1.0f);

  // Banded relative-k logits
  relk_kernel<<<(BH * LSEQ) / 256, 256, 0, stream>>>(Qf, erk, relk);

  // Flash attention with banded rel bias / rel-v
  attn_kernel<<<512, 256, 0, stream>>>(Qf, Kf, Vt, relk, erv, Xatt);

  // Output projection -> f32 d_out
  gemm_wmma<<<256, 256, 0, stream>>>(Xatt, Wto, bo, out, 0, 1.0f);
}
